// CausalSelfAttention_85985245265990
// MI455X (gfx1250) — matrix operand order repaired and checked by reading, not compiled, not run on hardware
//
#include <hip/hip_runtime.h>
#include <cstdint>
#include <cstddef>

#define N_HEADC 16
#define N_KVC   4
#define HEAD_D  128
#define D_MODEL (N_HEADC * HEAD_D)            // 2048
#define QKV_OUT ((N_HEADC + 2 * N_KVC) * HEAD_D) // 3072
#define BATCH   2
#define SEQ     2048
#define BTOK    (BATCH * SEQ)                 // 4096

typedef __bf16 bf16_t;
typedef __attribute__((ext_vector_type(16))) __bf16 v16bf;
typedef __attribute__((ext_vector_type(8)))  __bf16 v8bf;
typedef __attribute__((ext_vector_type(8)))  float  v8f;

__device__ __forceinline__ bf16_t f2bf(float f) {
  uint32_t u = __builtin_bit_cast(uint32_t, f);
  uint32_t r = u + 0x7FFFu + ((u >> 16) & 1u);   // round-to-nearest-even
  uint16_t s = (uint16_t)(r >> 16);
  return __builtin_bit_cast(bf16_t, s);
}

__device__ __forceinline__ v16bf cat16(v8bf lo, v8bf hi) {
  return __builtin_shufflevector(lo, hi, 0,1,2,3,4,5,6,7,8,9,10,11,12,13,14,15);
}

// CDNA5 async copy: global -> LDS, 16 bytes per lane, tracked by ASYNCcnt.
__device__ __forceinline__ void async_copy_b128(uint32_t lds_addr, const void* gaddr) {
  asm volatile("global_load_async_to_lds_b128 %0, %1, off"
               :: "v"(lds_addr), "v"(gaddr) : "memory");
}
__device__ __forceinline__ void wait_async0() {
  asm volatile("s_wait_asynccnt 0x0" ::: "memory");
}

// ---------------- conversion kernels ----------------

__global__ __launch_bounds__(256) void k_cvt(const float* __restrict__ in,
                                             unsigned short* __restrict__ outr, int n) {
  bf16_t* out = (bf16_t*)outr;
  int i = blockIdx.x * 256 + threadIdx.x;
  if (i < n) out[i] = f2bf(in[i]);
}

// in: [K,N] f32 row-major -> out: [N,K] bf16 row-major (B-operand friendly)
__global__ __launch_bounds__(256) void k_cvt_tr(const float* __restrict__ in,
                                                unsigned short* __restrict__ outr,
                                                int K, int N) {
  bf16_t* out = (bf16_t*)outr;
  int i = blockIdx.x * 256 + threadIdx.x;
  int k = i / N, n = i - k * N;
  out[(size_t)n * K + k] = f2bf(in[i]);
}

// ---------------- bf16 WMMA GEMM: C[M,N] = A[M,K] * Bt[N,K]^T ----------------
// Block tile 64(M) x 256(N); 8 waves in a 2x4 grid; wave tile 32x64 (2x4 WMMA tiles).
// A tile (64x32 bf16) double-buffered in LDS via async global->LDS copies.

#define GBM 64
#define GBN 256

__global__ __launch_bounds__(256) void k_gemm_bf16(const unsigned short* __restrict__ Ar_,
                                                   const unsigned short* __restrict__ Br_,
                                                   float* __restrict__ C,
                                                   int M, int N, int K) {
  const bf16_t* A  = (const bf16_t*)Ar_;
  const bf16_t* Bt = (const bf16_t*)Br_;
  __shared__ bf16_t Asm[2][GBM * 32];          // 2 x 4KB double buffer

  const int tid   = threadIdx.x;
  const int lane  = tid & 31;
  const int wave  = tid >> 5;
  const int wm    = wave & 1;                  // 2 M slots
  const int wn    = wave >> 1;                 // 4 N slots
  const int lrow  = lane & 15;
  const int lhalf = lane >> 4;

  const int m_blk = blockIdx.x * GBM;
  const int n0    = blockIdx.y * GBN + wn * 64;

  // staging: thread t copies row t/4, 8-element segment t%4 of the 64x32 A tile
  const int srow = tid >> 2, sseg = tid & 3;
  const bf16_t* Ag = A + (size_t)(m_blk + srow) * K + sseg * 8;
  const uint32_t lds0 = (uint32_t)(uintptr_t)(&Asm[0][0]) + (uint32_t)(srow * 64 + sseg * 16);
  const uint32_t ldsstride = GBM * 32 * 2;     // bytes per buffer

  const bf16_t* Brow[4];
#pragma unroll
  for (int nt = 0; nt < 4; ++nt)
    Brow[nt] = Bt + (size_t)(n0 + nt * 16 + lrow) * K;

  v8f acc[2][4];
#pragma unroll
  for (int mt = 0; mt < 2; ++mt)
#pragma unroll
    for (int nt = 0; nt < 4; ++nt) { v8f z = {}; acc[mt][nt] = z; }

  // prologue: stage kk=0 into buffer 0
  async_copy_b128(lds0, Ag);

  int p = 0;
  for (int kk = 0; kk < K; kk += 32) {
    wait_async0();           // my async stores to buf[p] landed
    __syncthreads();         // everyone's did
    if (kk + 32 < K)
      async_copy_b128(lds0 + (uint32_t)((p ^ 1) * ldsstride), Ag + kk + 32);

    // A fragments from LDS buffer p (WMMA 16-bit A layout)
    v16bf af[2];
#pragma unroll
    for (int mt = 0; mt < 2; ++mt) {
      const bf16_t* Ap = &Asm[p][(wm * 32 + mt * 16 + lrow) * 32 + lhalf * 8];
      v8bf lo = *(const v8bf*)Ap;
      v8bf hi = *(const v8bf*)(Ap + 16);
      af[mt] = cat16(lo, hi);
    }
    __builtin_prefetch(Brow[0] + kk + 256, 0, 1);
    // B fragments direct from global (L2-resident weights), 8 WMMA per k-step
#pragma unroll
    for (int nt = 0; nt < 4; ++nt) {
      v8bf b0 = *(const v8bf*)(Brow[nt] + kk + lhalf * 8);
      v8bf b1 = *(const v8bf*)(Brow[nt] + kk + lhalf * 8 + 16);
      v16bf bfr = cat16(b0, b1);
#pragma unroll
      for (int mt = 0; mt < 2; ++mt)
        acc[mt][nt] = __builtin_amdgcn_wmma_f32_16x16x32_bf16(false, af[mt], false, bfr,
                                                              (short)0, acc[mt][nt],
                                                              false, false);
    }
    __syncthreads();         // reads of buf[p] done before it is restaged
    p ^= 1;
  }

#pragma unroll
  for (int mt = 0; mt < 2; ++mt) {
#pragma unroll
    for (int nt = 0; nt < 4; ++nt) {
      float* Crow = C + (size_t)(m_blk + wm * 32 + mt * 16 + lhalf * 8) * N
                      + n0 + nt * 16 + lrow;
#pragma unroll
      for (int r = 0; r < 8; ++r) Crow[(size_t)r * N] = acc[mt][nt][r];
    }
  }
}

// ---------------- RMS-norm + pack q/k/v ----------------
// one wave per (b,t,unit); units: 16 q heads, 4 k heads, 4 v heads

__global__ __launch_bounds__(256) void k_norm_pack(const float* __restrict__ qkv,
                                                   const float* __restrict__ qs,
                                                   const float* __restrict__ ks,
                                                   unsigned short* __restrict__ qb_,
                                                   unsigned short* __restrict__ kb_,
                                                   unsigned short* __restrict__ vtb_) {
  bf16_t* qb  = (bf16_t*)qb_;
  bf16_t* kb  = (bf16_t*)kb_;
  bf16_t* vtb = (bf16_t*)vtb_;
  const int lane = threadIdx.x & 31;
  int u    = blockIdx.x * 8 + (threadIdx.x >> 5);
  int bt   = u / (N_HEADC + 2 * N_KVC);
  int unit = u - bt * (N_HEADC + 2 * N_KVC);
  int b = bt / SEQ, t = bt - b * SEQ;
  const float* row = qkv + (size_t)bt * QKV_OUT;

  if (unit < N_HEADC + N_KVC) {
    const bool isq = unit < N_HEADC;
    int head = isq ? unit : (unit - N_HEADC);
    const float* src = isq ? (row + head * HEAD_D)
                           : (row + D_MODEL + head * HEAD_D);
    float vals[4];
    float ss = 0.f;
#pragma unroll
    for (int i = 0; i < 4; ++i) { vals[i] = src[lane + 32 * i]; ss += vals[i] * vals[i]; }
#pragma unroll
    for (int m = 1; m < 32; m <<= 1) ss += __shfl_xor(ss, m, 32);
    float inv = rsqrtf(ss * (1.f / HEAD_D) + 1e-6f);
    if (isq) {
      bf16_t* dst = qb + ((size_t)(b * N_HEADC + head) * SEQ + t) * HEAD_D;
#pragma unroll
      for (int i = 0; i < 4; ++i) { int d = lane + 32 * i; dst[d] = f2bf(vals[i] * inv * qs[d]); }
    } else {
      bf16_t* dst = kb + ((size_t)(b * N_KVC + head) * SEQ + t) * HEAD_D;
#pragma unroll
      for (int i = 0; i < 4; ++i) { int d = lane + 32 * i; dst[d] = f2bf(vals[i] * inv * ks[d]); }
    }
  } else {
    int kv = unit - N_HEADC - N_KVC;
    const float* src = row + D_MODEL + N_KVC * HEAD_D + kv * HEAD_D;
#pragma unroll
    for (int i = 0; i < 4; ++i) {
      int d = lane + 32 * i;   // vT layout: [B,KV,D,T], T contiguous
      vtb[((size_t)(b * N_KVC + kv) * HEAD_D + d) * SEQ + t] = f2bf(src[d]);
    }
  }
}

// ---------------- flash attention, one 16-row query tile per wave ----------------

__global__ __launch_bounds__(128) void k_flash(const unsigned short* __restrict__ qb_,
                                               const unsigned short* __restrict__ kb_,
                                               const unsigned short* __restrict__ vtb_,
                                               unsigned short* __restrict__ yb_) {
  const bf16_t* qbuf = (const bf16_t*)qb_;
  const bf16_t* kbuf = (const bf16_t*)kb_;
  const bf16_t* vtb  = (const bf16_t*)vtb_;
  bf16_t*       ybuf = (bf16_t*)yb_;

  __shared__ bf16_t plds_all[4][16 * 32];   // per-wave P scratch (16q x 32key)
  const int lane  = threadIdx.x & 31;
  const int wave  = threadIdx.x >> 5;
  bf16_t* P = plds_all[wave];

  int job = blockIdx.x * 4 + wave;          // 4096 jobs total
  int qt  = job & (SEQ / 16 - 1);           // 128 q-tiles
  int bh  = job >> 7;
  int h   = bh & (N_HEADC - 1);
  int b   = bh >> 4;
  int kv  = h >> 2;                         // GQA: 4 q-heads per kv-head
  int q0  = qt * 16;
  const int lrow = lane & 15, lhalf = lane >> 4;

  // Q fragments: 16x128 as 4 A-operand chunks
  v16bf qf[4];
  {
    const bf16_t* Qr = qbuf + ((size_t)(b * N_HEADC + h) * SEQ + q0 + lrow) * HEAD_D;
#pragma unroll
    for (int c = 0; c < 4; ++c) {
      v8bf lo = *(const v8bf*)(Qr + c * 32 + lhalf * 8);
      v8bf hi = *(const v8bf*)(Qr + c * 32 + lhalf * 8 + 16);
      qf[c] = cat16(lo, hi);
    }
  }

  v8f acc[8];
#pragma unroll
  for (int i = 0; i < 8; ++i) { v8f z = {}; acc[i] = z; }
  float rowm[8], rowl[8];
#pragma unroll
  for (int r = 0; r < 8; ++r) { rowm[r] = -3.0e38f; rowl[r] = 0.f; }

  const bf16_t* Kbase = kbuf + (size_t)(b * N_KVC + kv) * SEQ * HEAD_D;
  const bf16_t* Vbase = vtb  + (size_t)(b * N_KVC + kv) * HEAD_D * SEQ;
  const float sscale = 0.088388347648318447f;  // 1/sqrt(128)

  const int kend = q0 + 16;
  for (int j = 0; j < kend; j += 32) {
    // ---- scores: two 16-key subtiles, contraction over D in 4 chunks ----
    v8f S[2];
#pragma unroll
    for (int s = 0; s < 2; ++s) {
      v8f sc = {};
      const bf16_t* Kr = Kbase + (size_t)(j + s * 16 + lrow) * HEAD_D;
#pragma unroll
      for (int c = 0; c < 4; ++c) {
        v8bf b0 = *(const v8bf*)(Kr + c * 32 + lhalf * 8);
        v8bf b1 = *(const v8bf*)(Kr + c * 32 + lhalf * 8 + 16);
        sc = __builtin_amdgcn_wmma_f32_16x16x32_bf16(false, qf[c], false, cat16(b0, b1),
                                                     (short)0, sc, false, false);
      }
      S[s] = sc;
    }
    // ---- scale + causal mask (branchless, per-lane select) ----
#pragma unroll
    for (int s = 0; s < 2; ++s) {
      int kn = j + s * 16 + lrow;
#pragma unroll
      for (int r = 0; r < 8; ++r) {
        int qr = q0 + lhalf * 8 + r;
        float v = S[s][r] * sscale;
        S[s][r] = (kn <= qr) ? v : -3.0e38f;
      }
    }
    // ---- online softmax: row reductions across the 16-lane half ----
    float alpha[8], mnew[8];
#pragma unroll
    for (int r = 0; r < 8; ++r) {
      float rm = fmaxf(S[0][r], S[1][r]);
#pragma unroll
      for (int m = 1; m < 16; m <<= 1) rm = fmaxf(rm, __shfl_xor(rm, m, 32));
      mnew[r]  = fmaxf(rowm[r], rm);
      alpha[r] = __expf(rowm[r] - mnew[r]);
      rowm[r]  = mnew[r];
    }
#pragma unroll
    for (int r = 0; r < 8; ++r) {
      float p0 = __expf(S[0][r] - mnew[r]);
      float p1 = __expf(S[1][r] - mnew[r]);
      float ps = p0 + p1;
#pragma unroll
      for (int m = 1; m < 16; m <<= 1) ps += __shfl_xor(ps, m, 32);
      rowl[r] = rowl[r] * alpha[r] + ps;
      // spill P (D-layout) into LDS row-major [16][32]
      P[(lhalf * 8 + r) * 32 + lrow]      = f2bf(p0);
      P[(lhalf * 8 + r) * 32 + 16 + lrow] = f2bf(p1);
    }
    // ---- rescale accumulators ----
#pragma unroll
    for (int od = 0; od < 8; ++od)
#pragma unroll
      for (int r = 0; r < 8; ++r) acc[od][r] *= alpha[r];
    // ---- reload P as A-operand (LDS transpose bounce) ----
    v16bf pf;
    {
      v8bf lo = *(const v8bf*)(P + lrow * 32 + lhalf * 8);
      v8bf hi = *(const v8bf*)(P + lrow * 32 + 16 + lhalf * 8);
      pf = cat16(lo, hi);
    }
    // ---- P(16x32) @ V(32x128): 8 WMMA ----
#pragma unroll
    for (int od = 0; od < 8; ++od) {
      const bf16_t* Vr = Vbase + (size_t)(od * 16 + lrow) * SEQ + j;
      v8bf b0 = *(const v8bf*)(Vr + lhalf * 8);
      v8bf b1 = *(const v8bf*)(Vr + lhalf * 8 + 16);
      acc[od] = __builtin_amdgcn_wmma_f32_16x16x32_bf16(false, pf, false, cat16(b0, b1),
                                                        (short)0, acc[od], false, false);
    }
  }

  // ---- normalize and write y[b,t, h*D + d] as bf16 ----
#pragma unroll
  for (int r = 0; r < 8; ++r) {
    float inv = 1.f / rowl[r];
    int qr = q0 + lhalf * 8 + r;
    bf16_t* Yr = ybuf + ((size_t)(b * SEQ + qr)) * D_MODEL + h * HEAD_D;
#pragma unroll
    for (int od = 0; od < 8; ++od) Yr[od * 16 + lrow] = f2bf(acc[od][r] * inv);
  }
}

// ---------------- host launcher ----------------

extern "C" void kernel_launch(void* const* d_in, const int* in_sizes, int n_in,
                              void* d_out, int out_size, void* d_ws, size_t ws_size,
                              hipStream_t stream) {
  (void)in_sizes; (void)n_in; (void)out_size; (void)ws_size;
  const float* x      = (const float*)d_in[0];
  const float* W_qkv  = (const float*)d_in[1];
  const float* W_proj = (const float*)d_in[2];
  const float* q_scale = (const float*)d_in[3];
  const float* k_scale = (const float*)d_in[4];
  float* out = (float*)d_out;

  char* w = (char*)d_ws;
  unsigned short* xb     = (unsigned short*)w; w += (size_t)BTOK * D_MODEL * 2;
  unsigned short* wqkvT  = (unsigned short*)w; w += (size_t)QKV_OUT * D_MODEL * 2;
  unsigned short* wprojT = (unsigned short*)w; w += (size_t)D_MODEL * D_MODEL * 2;
  float*          qkvf   = (float*)w;          w += (size_t)BTOK * QKV_OUT * 4;
  unsigned short* qb     = (unsigned short*)w; w += (size_t)BATCH * N_HEADC * SEQ * HEAD_D * 2;
  unsigned short* kb     = (unsigned short*)w; w += (size_t)BATCH * N_KVC * SEQ * HEAD_D * 2;
  unsigned short* vtb    = (unsigned short*)w; w += (size_t)BATCH * N_KVC * SEQ * HEAD_D * 2;
  unsigned short* yb     = (unsigned short*)w; w += (size_t)BTOK * D_MODEL * 2;

  // 1) convert inputs to bf16 (weights transposed for B-operand layout)
  k_cvt<<<(BTOK * D_MODEL) / 256, 256, 0, stream>>>(x, xb, BTOK * D_MODEL);
  k_cvt_tr<<<(D_MODEL * QKV_OUT) / 256, 256, 0, stream>>>(W_qkv, wqkvT, D_MODEL, QKV_OUT);
  k_cvt_tr<<<(D_MODEL * D_MODEL) / 256, 256, 0, stream>>>(W_proj, wprojT, D_MODEL, D_MODEL);

  // 2) qkv = x @ W_qkv  (f32 accum)
  k_gemm_bf16<<<dim3(BTOK / GBM, QKV_OUT / GBN), 256, 0, stream>>>(
      xb, wqkvT, qkvf, BTOK, QKV_OUT, D_MODEL);

  // 3) RMS-norm q/k + pack v (transposed)
  k_norm_pack<<<(BTOK * (N_HEADC + 2 * N_KVC)) / 8, 256, 0, stream>>>(
      qkvf, q_scale, k_scale, qb, kb, vtb);

  // 4) causal flash attention
  k_flash<<<(BATCH * N_HEADC * (SEQ / 16)) / 4, 128, 0, stream>>>(qb, kb, vtb, yb);

  // 5) out = y @ W_proj (f32 result)
  k_gemm_bf16<<<dim3(BTOK / GBM, D_MODEL / GBN), 256, 0, stream>>>(
      yb, wprojT, out, BTOK, D_MODEL, D_MODEL);
}
